// CalibrationModelObsGridGeometry_25142738550839
// MI455X (gfx1250) — matrix operand order repaired and checked by reading, compiled, not run on hardware
//
#include <hip/hip_runtime.h>
#include <hip/hip_bf16.h>

typedef __attribute__((ext_vector_type(16))) _Float16 v16h;
typedef __attribute__((ext_vector_type(8)))  _Float16 v8h;
typedef __attribute__((ext_vector_type(8)))  float    v8f;

#define HI_     1100
#define W_      52
#define M_      24
#define NPIX    (M_ * HI_ * W_)       // 1,372,800
#define HFULL   1200
#define KSIZE   75
#define KHALF   37
#define NSIG    10
#define NPLANE  12                    // unique BN channels
#define OUTPLANE (32 * HI_ * W_)      // 1,830,400 (one output tensor)
#define KPAD1   224                   // 22*9=198 padded to 7 chunks of 32
#define NQC1    7
#define KPAD2   288                   // 32*9 exactly, 9 chunks
#define NQC2    9
#define BN_EPS_ 1e-5f

// ---------------------------------------------------------------- init
__global__ void k_init(float* out, float* stats) {
    int i = blockIdx.x * blockDim.x + threadIdx.x;
    int tot = 2 * OUTPLANE;
    if (i < tot) out[i] = 0.0f;
    if (i < 64)  stats[i] = 0.0f;
}

// ---------------------------------------------------------------- gather
__global__ void k_gather(const float* __restrict__ sv_uncal,
                         const float* __restrict__ sv_bg,
                         const int* __restrict__ msk_idx,
                         const int* __restrict__ row_idx,
                         float* __restrict__ fy_sel,
                         float* __restrict__ fs_sel) {
    int p = blockIdx.x * blockDim.x + threadIdx.x;
    if (p >= NPIX) return;
    int w  = p % W_;
    int hi = (p / W_) % HI_;
    int m  = p / (W_ * HI_);
    int src = msk_idx[m] * (HFULL * W_) + row_idx[hi] * W_ + w;
    fy_sel[p] = sv_uncal[src];
    fs_sel[p] = sv_bg[src];
}

// ------------------------------------------- gaussian bank + diff + BN stats
__global__ __launch_bounds__(256)
void k_gauss(const float* __restrict__ fy, const float* __restrict__ fs,
             const float* __restrict__ gk, _Float16* __restrict__ cal,
             float* __restrict__ stats) {
    __shared__ float sgk[NSIG * KSIZE];
    __shared__ float sred[2 * NPLANE];
    int tid = threadIdx.x;
    for (int i = tid; i < NSIG * KSIZE; i += blockDim.x) sgk[i] = gk[i];
    if (tid < 2 * NPLANE) sred[tid] = 0.0f;
    __syncthreads();

    int p = blockIdx.x * blockDim.x + tid;
    float vals[NPLANE];
#pragma unroll
    for (int c = 0; c < NPLANE; ++c) vals[c] = 0.0f;

    if (p < NPIX) {
        int w    = p % W_;
        int hi   = (p / W_) % HI_;
        int m    = p / (W_ * HI_);
        int base = m * HI_ * W_ + w;
        float acc[NSIG];
#pragma unroll
        for (int s = 0; s < NSIG; ++s) acc[s] = 0.0f;
        float accbg = 0.0f;
        for (int k = 0; k < KSIZE; ++k) {
            int hh = hi + k - KHALF;
            hh = hh < 0 ? 0 : (hh >= HI_ ? HI_ - 1 : hh);
            float xy = fy[base + hh * W_];
            float xs = fs[base + hh * W_];
#pragma unroll
            for (int s = 0; s < NSIG; ++s) acc[s] += xy * sgk[s * KSIZE + k];
            accbg += xs * sgk[(NSIG - 1) * KSIZE + k];
        }
        float g0 = fy[p];
        vals[0] = g0 - acc[0];
#pragma unroll
        for (int c = 1; c < 10; ++c) vals[c] = acc[c - 1] - acc[c];
        vals[10] = acc[NSIG - 1];
        vals[11] = accbg;
#pragma unroll
        for (int c = 0; c < NPLANE; ++c)
            cal[c * NPIX + p] = (_Float16)vals[c];
    }
#pragma unroll
    for (int c = 0; c < NPLANE; ++c) {
        atomicAdd(&sred[c], vals[c]);
        atomicAdd(&sred[NPLANE + c], vals[c] * vals[c]);
    }
    __syncthreads();
    if (tid < 2 * NPLANE) atomicAdd(&stats[tid], sred[tid]);
}

// ---------------------------------------------------------------- BN finalize
__global__ void k_bnfin(float* stats) {
    int t = threadIdx.x;
    if (t < NPLANE) {
        float inv_n = 1.0f / (float)NPIX;
        float mean  = stats[t] * inv_n;
        float var   = stats[NPLANE + t] * inv_n - mean * mean;
        stats[24 + t] = mean;
        stats[36 + t] = rsqrtf(var + BN_EPS_);
    }
}

__device__ __forceinline__ int ucmap(int c) {
    return (c <= 10) ? c : ((c == 21) ? 11 : c - 11);
}

// ---------------------------------------------------------------- conv1 WMMA
// BN folded into weights: y = x (*) (w * r_c)  +  (b - sum w*mu_c*r_c)
__global__ __launch_bounds__(256)
void k_conv1(const _Float16* __restrict__ cal, const float* __restrict__ stats,
             const float* __restrict__ w1, const float* __restrict__ b1,
             _Float16* __restrict__ h1o) {
    __shared__ _Float16 As[8 * NQC1 * 512];   // 8 tiles x 7 chunks x (32 lanes x 16 halfs)
    __shared__ _Float16 Bf[2 * NQC1 * 512];   // 2 n-tiles
    __shared__ float smean[NPLANE], srstd[NPLANE];
    __shared__ float sbias[32];
    int tid = threadIdx.x;

    if (tid < NPLANE) { smean[tid] = stats[24 + tid]; srstd[tid] = stats[36 + tid]; }
    __syncthreads();

    // B fragments (BN scale folded in), fragment-major layout
    for (int i = tid; i < 2 * NQC1 * 512; i += 256) {
        int j    = i & 15;
        int lane = (i >> 4) & 31;
        int rest = i >> 9;
        int qc   = rest % NQC1;
        int nt   = rest / NQC1;
        int k    = qc * 32 + ((j >> 3) << 4) + ((lane >> 4) << 3) + (j & 7);
        int n    = (nt << 4) + (lane & 15);
        float v = 0.0f;
        if (k < 198) v = w1[n * 198 + k] * srstd[ucmap(k / 9)];
        Bf[i] = (_Float16)v;
    }
    // adjusted bias
    if (tid < 32) {
        float s = b1[tid];
        int c = 0, t = 0;
        for (int k = 0; k < 198; ++k) {
            int uc = ucmap(c);
            s -= w1[tid * 198 + k] * smean[uc] * srstd[uc];
            if (++t == 9) { t = 0; ++c; }
        }
        sbias[tid] = s;
    }
    // A fragments: raw cal values, im2col, fragment-major layout
    {
        int px   = tid & 127;
        int half = tid >> 7;
        int tile = px >> 4;
        int m    = px & 15;
        int p    = blockIdx.x * 128 + px;
        int ww   = p % W_;
        int hh   = (p / W_) % HI_;
        int c = 0, t = half;
        for (int r = half; r < KPAD1; r += 2) {
            float v = 0.0f;
            if (c < 22) {
                int dy = (t >= 6) ? 1 : ((t >= 3) ? 0 : -1);
                int dx = t - (dy + 1) * 3 - 1;
                int h2i = hh + dy, w2i = ww + dx;
                if (h2i >= 0 && h2i < HI_ && w2i >= 0 && w2i < W_)
                    v = (float)cal[ucmap(c) * NPIX + p + dy * W_ + dx];
            }
            int qc = r >> 5, kl = r & 31;
            int g  = (kl >> 3) & 1;
            int j  = ((kl >> 4) << 3) | (kl & 7);
            As[((tile * NQC1 + qc) << 9) + (((g << 4) | m) << 4) + j] = (_Float16)v;
            t += 2; if (t >= 9) { t -= 9; ++c; }
        }
    }
    __syncthreads();

    int wave = tid >> 5, lane = tid & 31;
    int g = lane >> 4, mr = lane & 15;
    v8f acc0 = {}, acc1 = {};
#pragma unroll
    for (int qc = 0; qc < NQC1; ++qc) {
        v16h A  = *(const v16h*)&As[((wave * NQC1 + qc) << 9) + (lane << 4)];
        v16h B0 = *(const v16h*)&Bf[(qc << 9) + (lane << 4)];
        v16h B1 = *(const v16h*)&Bf[((NQC1 + qc) << 9) + (lane << 4)];
        acc0 = __builtin_amdgcn_wmma_f32_16x16x32_f16(false, A, false, B0, (short)0, acc0, false, false);
        acc1 = __builtin_amdgcn_wmma_f32_16x16x32_f16(false, A, false, B1, (short)0, acc1, false, false);
    }
    int tbase = blockIdx.x * 128 + wave * 16;
    v8h o0, o1;
#pragma unroll
    for (int i = 0; i < 8; ++i) {
        float v0 = acc0[i] + sbias[mr];       v0 = v0 > 0.0f ? v0 : 0.0f;
        float v1 = acc1[i] + sbias[16 + mr];  v1 = v1 > 0.0f ? v1 : 0.0f;
        o0[i] = (_Float16)v0;
        o1[i] = (_Float16)v1;
    }
    *(v8h*)&h1o[mr * NPIX + tbase + (g << 3)]        = o0;
    *(v8h*)&h1o[(16 + mr) * NPIX + tbase + (g << 3)] = o1;
}

// ---------------------------------------------------------------- conv2 WMMA
__global__ __launch_bounds__(256)
void k_conv2(const _Float16* __restrict__ h1, const float* __restrict__ w2,
             const float* __restrict__ b2, _Float16* __restrict__ h2o) {
    __shared__ _Float16 As[8 * NQC2 * 512];
    __shared__ _Float16 Bf[2 * NQC2 * 512];
    int tid = threadIdx.x;

    for (int i = tid; i < 2 * NQC2 * 512; i += 256) {
        int j    = i & 15;
        int lane = (i >> 4) & 31;
        int rest = i >> 9;
        int qc   = rest % NQC2;
        int nt   = rest / NQC2;
        int k    = qc * 32 + ((j >> 3) << 4) + ((lane >> 4) << 3) + (j & 7);
        int n    = (nt << 4) + (lane & 15);
        Bf[i] = (_Float16)w2[n * 288 + k];
    }
    {
        int px   = tid & 127;
        int half = tid >> 7;
        int tile = px >> 4;
        int m    = px & 15;
        int p    = blockIdx.x * 128 + px;
        int ww   = p % W_;
        int hh   = (p / W_) % HI_;
        int c = 0, t = half;
        for (int r = half; r < KPAD2; r += 2) {
            int dy = (t >= 6) ? 1 : ((t >= 3) ? 0 : -1);
            int dx = t - (dy + 1) * 3 - 1;
            int h2i = hh + dy, w2i = ww + dx;
            _Float16 v = (_Float16)0.0f;
            if (h2i >= 0 && h2i < HI_ && w2i >= 0 && w2i < W_)
                v = h1[c * NPIX + p + dy * W_ + dx];
            int qc = r >> 5, kl = r & 31;
            int g  = (kl >> 3) & 1;
            int j  = ((kl >> 4) << 3) | (kl & 7);
            As[((tile * NQC2 + qc) << 9) + (((g << 4) | m) << 4) + j] = v;
            t += 2; if (t >= 9) { t -= 9; ++c; }
        }
    }
    __syncthreads();

    int wave = tid >> 5, lane = tid & 31;
    int g = lane >> 4, mr = lane & 15;
    v8f acc0 = {}, acc1 = {};
#pragma unroll
    for (int qc = 0; qc < NQC2; ++qc) {
        v16h A  = *(const v16h*)&As[((wave * NQC2 + qc) << 9) + (lane << 4)];
        v16h B0 = *(const v16h*)&Bf[(qc << 9) + (lane << 4)];
        v16h B1 = *(const v16h*)&Bf[((NQC2 + qc) << 9) + (lane << 4)];
        acc0 = __builtin_amdgcn_wmma_f32_16x16x32_f16(false, A, false, B0, (short)0, acc0, false, false);
        acc1 = __builtin_amdgcn_wmma_f32_16x16x32_f16(false, A, false, B1, (short)0, acc1, false, false);
    }
    int tbase = blockIdx.x * 128 + wave * 16;
    v8h o0, o1;
#pragma unroll
    for (int i = 0; i < 8; ++i) {
        float v0 = acc0[i] + b2[mr];       v0 = v0 > 0.0f ? v0 : 0.0f;
        float v1 = acc1[i] + b2[16 + mr];  v1 = v1 > 0.0f ? v1 : 0.0f;
        o0[i] = (_Float16)v0;
        o1[i] = (_Float16)v1;
    }
    *(v8h*)&h2o[mr * NPIX + tbase + (g << 3)]        = o0;
    *(v8h*)&h2o[(16 + mr) * NPIX + tbase + (g << 3)] = o1;
}

// -------------------------------------------- conv3 + epilogue + scatter + mask
__global__ __launch_bounds__(256)
void k_conv3(const _Float16* __restrict__ h2, const float* __restrict__ w3,
             const float* __restrict__ b3, const float* __restrict__ fs_sel,
             const int* __restrict__ msk_idx, float* __restrict__ out) {
    __shared__ float sw[32 * 9];
    int tid = threadIdx.x;
    for (int i = tid; i < 32 * 9; i += blockDim.x) sw[i] = w3[i];
    __syncthreads();

    int p = blockIdx.x * blockDim.x + tid;
    if (p >= NPIX) return;
    int ww = p % W_;
    int hh = (p / W_) % HI_;
    int mm = p / (W_ * HI_);

    float acc = b3[0];
#pragma unroll
    for (int t = 0; t < 9; ++t) {
        int dy = t / 3 - 1, dx = t % 3 - 1;
        int h2i = hh + dy, w2i = ww + dx;
        if (h2i < 0 || h2i >= HI_ || w2i < 0 || w2i >= W_) continue;
        int pn = p + dy * W_ + dx;
#pragma unroll
        for (int c = 0; c < 32; ++c)
            acc += (float)h2[c * NPIX + pn] * sw[c * 9 + t];
    }
    const float nsc = (float)(0.31446309894037083 / 0.3886609494201447);
    float o = acc + nsc + fs_sel[p];
    int mf = msk_idx[mm];
    int oidx = mf * (HI_ * W_) + hh * W_ + ww;
    out[oidx] = o;
    out[OUTPLANE + oidx] = 1.0f;
}

// ---------------------------------------------------------------- launch
extern "C" void kernel_launch(void* const* d_in, const int* in_sizes, int n_in,
                              void* d_out, int out_size, void* d_ws, size_t ws_size,
                              hipStream_t stream) {
    const float* sv_uncal = (const float*)d_in[0];
    const float* sv_bg    = (const float*)d_in[1];
    const float* gkern    = (const float*)d_in[2];
    const float* w1       = (const float*)d_in[3];
    const float* b1       = (const float*)d_in[4];
    const float* w2       = (const float*)d_in[5];
    const float* b2       = (const float*)d_in[6];
    const float* w3       = (const float*)d_in[7];
    const float* b3       = (const float*)d_in[8];
    const int*   msk_idx  = (const int*)d_in[9];
    const int*   row_idx  = (const int*)d_in[10];
    float* out = (float*)d_out;

    char* ws = (char*)d_ws;
    size_t off = 0;
    float*    fy_sel = (float*)(ws + off);    off += (size_t)NPIX * 4;
    float*    fs_sel = (float*)(ws + off);    off += (size_t)NPIX * 4;
    float*    stats  = (float*)(ws + off);    off += 256;
    _Float16* cal    = (_Float16*)(ws + off); off += (size_t)NPLANE * NPIX * 2;
    _Float16* h1buf  = (_Float16*)(ws + off); off += (size_t)32 * NPIX * 2;
    _Float16* h2buf  = (_Float16*)(ws + off); off += (size_t)32 * NPIX * 2;

    int tpix  = (NPIX + 255) / 256;
    int tinit = (2 * OUTPLANE + 255) / 256;
    int tconv = NPIX / 128;   // 10,725 blocks, 8 wave-tiles each

    k_init  <<<tinit, 256, 0, stream>>>(out, stats);
    k_gather<<<tpix, 256, 0, stream>>>(sv_uncal, sv_bg, msk_idx, row_idx, fy_sel, fs_sel);
    k_gauss <<<tpix, 256, 0, stream>>>(fy_sel, fs_sel, gkern, cal, stats);
    k_bnfin <<<1, 32, 0, stream>>>(stats);
    k_conv1 <<<tconv, 256, 0, stream>>>(cal, stats, w1, b1, h1buf);
    k_conv2 <<<tconv, 256, 0, stream>>>(h1buf, w2, b2, h2buf);
    k_conv3 <<<tpix, 256, 0, stream>>>(h2buf, w3, b3, fs_sel, msk_idx, out);
}
